// CrossAttention_7756710936941
// MI455X (gfx1250) — matrix operand order repaired and checked by reading, not compiled, not run on hardware
//
#include <hip/hip_runtime.h>
#include <hip/hip_bf16.h>

typedef __attribute__((ext_vector_type(16))) __bf16 v16bf;
typedef __attribute__((ext_vector_type(8)))  __bf16 v8bf;
typedef __attribute__((ext_vector_type(4)))  __bf16 v4bf;
typedef __attribute__((ext_vector_type(2)))  __bf16 v2bf;
typedef __attribute__((ext_vector_type(8)))  float  v8f;
typedef __attribute__((ext_vector_type(4)))  unsigned int v4u;
typedef __attribute__((ext_vector_type(8)))  int v8i_;
typedef __attribute__((ext_vector_type(4)))  int v4i_;

// ---------------------------------------------------------------------------
// WMMA helper: D = A(16x32 bf16) * B(32x16 bf16) + C(16x16 f32)
// ---------------------------------------------------------------------------
__device__ __forceinline__ v8f wmma_bf16(v16bf a, v16bf b, v8f c) {
  return __builtin_amdgcn_wmma_f32_16x16x32_bf16(
      false, a, false, b, (short)0, c, false, false);
}

// A-fragment loader (16x32, MxK) from row-major LDS tile.
__device__ __forceinline__ v16bf lds_load_a16(const __bf16* base, int ldk) {
  int lane = threadIdx.x & 31;
  const __bf16* p = base + (lane & 15) * ldk + ((lane >> 4) << 3);
  v8bf lo = *(const v8bf*)(p);
  v8bf hi = *(const v8bf*)(p + 16);
  return __builtin_shufflevector(lo, hi, 0,1,2,3,4,5,6,7,8,9,10,11,12,13,14,15);
}

// B-fragment loader (32x16, KxN) from N-major (transposed) LDS tile [n][k].
__device__ __forceinline__ v16bf lds_load_b16(const __bf16* base, int ldk) {
  int lane = threadIdx.x & 31;
  const __bf16* p = base + (lane & 15) * ldk + ((lane >> 4) << 3);
  v8bf lo = *(const v8bf*)(p);
  v8bf hi = *(const v8bf*)(p + 16);
  return __builtin_shufflevector(lo, hi, 0,1,2,3,4,5,6,7,8,9,10,11,12,13,14,15);
}

// Async LDS DMA (CDNA5): copies 16 bytes per lane, tracked by ASYNCcnt.
__device__ __forceinline__ void async_load_b128(unsigned lds_byte_off,
                                                const void* gptr) {
  asm volatile("global_load_async_to_lds_b128 %0, %1, off"
               :: "v"(lds_byte_off), "v"(gptr) : "memory");
}
__device__ __forceinline__ void wait_asynccnt0() {
  asm volatile("s_wait_asynccnt 0x0" ::: "memory");
}

// Tensor Data Mover: 2D tile (rows x cols bf16) global -> LDS with LDS row
// padding. Tracked by TENSORcnt.
__device__ __forceinline__ void tdm_load_2d_bf16(unsigned lds_off,
                                                 const void* gaddr,
                                                 int tile_cols, int tile_rows,
                                                 int row_stride_elems,
                                                 int pad_interval_code,
                                                 int pad_amount_code) {
  unsigned long long ga = (unsigned long long)(uintptr_t)gaddr;
  v4u g0;
  g0[0] = 1u;                                   // count=1, user mode
  g0[1] = lds_off;                              // lds_addr
  g0[2] = (unsigned)ga;                         // global_addr[31:0]
  g0[3] = (unsigned)(ga >> 32) | (2u << 30);    // global_addr[56:32] | type=2
  v8i_ g1;
  g1[0] = (int)((1u << 16)                      // data_size = 2 bytes
              | (1u << 20)                      // pad_enable
              | ((unsigned)pad_interval_code << 22)
              | ((unsigned)pad_amount_code << 25));
  g1[1] = (int)(((unsigned)tile_cols & 0xFFFFu) << 16);   // tensor_dim0 lo
  g1[2] = (int)(((unsigned)tile_rows & 0xFFFFu) << 16);   // tensor_dim1 lo
  g1[3] = (int)(((unsigned)tile_cols & 0xFFFFu) << 16);   // tile_dim0
  g1[4] = (int)((unsigned)tile_rows & 0xFFFFu);           // tile_dim1
  g1[5] = row_stride_elems;                               // tensor_dim0_stride lo
  g1[6] = 0;
  g1[7] = 0;
  v4i_ gz = {0, 0, 0, 0};
#if defined(__clang_major__) && (__clang_major__ >= 23)
  v8i_ gz8 = {0, 0, 0, 0, 0, 0, 0, 0};
  __builtin_amdgcn_tensor_load_to_lds(g0, g1, gz, gz, gz8, 0);
#else
  __builtin_amdgcn_tensor_load_to_lds(g0, g1, gz, gz, 0);
#endif
}

// ---------------------------------------------------------------------------
// GEMM tiles: 64(M) x 128(N) per block, K-step 32, 8 waves -> 32x32 per wave.
// Double-buffered LDS software pipeline.
// ---------------------------------------------------------------------------
constexpr int G_BM = 64;
constexpr int G_BN = 128;
constexpr int G_BK = 32;
constexpr int G_LDA = G_BK + 8;   // 40 elems
constexpr int G_LDW = G_BK + 8;

__device__ __forceinline__ void load_w_regs(const float* __restrict__ W,
                                            int k0, int n0, int N,
                                            int seg, int kp, float4 f[4]) {
  #pragma unroll
  for (int g = 0; g < 2; ++g) {
    int p = kp + g * 8;                               // pair index
    f[2*g+0] = *(const float4*)&W[(size_t)(k0 + 2*p + 0) * N + n0 + seg * 4];
    f[2*g+1] = *(const float4*)&W[(size_t)(k0 + 2*p + 1) * N + n0 + seg * 4];
  }
}
__device__ __forceinline__ void store_w_lds(__bf16* Ws, int seg, int kp,
                                            const float4 f[4]) {
  #pragma unroll
  for (int g = 0; g < 2; ++g) {
    int p = kp + g * 8;
    const float4& a = f[2*g+0];
    const float4& b = f[2*g+1];
    v2bf p0 = { (__bf16)a.x, (__bf16)b.x };
    v2bf p1 = { (__bf16)a.y, (__bf16)b.y };
    v2bf p2 = { (__bf16)a.z, (__bf16)b.z };
    v2bf p3 = { (__bf16)a.w, (__bf16)b.w };
    *(v2bf*)&Ws[(seg * 4 + 0) * G_LDW + 2 * p] = p0;
    *(v2bf*)&Ws[(seg * 4 + 1) * G_LDW + 2 * p] = p1;
    *(v2bf*)&Ws[(seg * 4 + 2) * G_LDW + 2 * p] = p2;
    *(v2bf*)&Ws[(seg * 4 + 3) * G_LDW + 2 * p] = p3;
  }
}
__device__ __forceinline__ void gemm_compute(const __bf16* As, const __bf16* Ws,
                                             int wm, int wn, v8f acc[2][2]) {
  v16bf a0 = lds_load_a16(&As[(wm +  0) * G_LDA], G_LDA);
  v16bf a1 = lds_load_a16(&As[(wm + 16) * G_LDA], G_LDA);
  v16bf b0 = lds_load_b16(&Ws[(wn +  0) * G_LDW], G_LDW);
  v16bf b1 = lds_load_b16(&Ws[(wn + 16) * G_LDW], G_LDW);
  acc[0][0] = wmma_bf16(a0, b0, acc[0][0]);
  acc[0][1] = wmma_bf16(a0, b1, acc[0][1]);
  acc[1][0] = wmma_bf16(a1, b0, acc[1][0]);
  acc[1][1] = wmma_bf16(a1, b1, acc[1][1]);
}

// C(bf16) = A(f32) * W(f32) + bias
__global__ __launch_bounds__(256)
void gemm_qkv(const float* __restrict__ A, const float* __restrict__ W,
              const float* __restrict__ bias, __bf16* __restrict__ C,
              int M, int N, int K) {
  __shared__ __bf16 As[2][G_BM * G_LDA];
  __shared__ __bf16 Ws[2][G_BN * G_LDW];
  int tid  = threadIdx.x;
  int wave = tid >> 5;
  int lane = tid & 31;
  int m0 = blockIdx.y * G_BM;
  int n0 = blockIdx.x * G_BN;
  int wm = (wave >> 2) * 32;
  int wn = (wave & 3) * 32;

  int aquad = tid & 7;           // float4 within a 32-wide A row
  int arow  = tid >> 3;          // 0..31 (+32 second half)
  int wseg  = tid & 31;
  int wkp   = tid >> 5;          // 0..7

  v8f acc[2][2] = {};

  // prologue: stage tile 0 into buffer 0
  {
    float4 fa0 = *(const float4*)&A[(size_t)(m0 + arow)      * K + aquad * 4];
    float4 fa1 = *(const float4*)&A[(size_t)(m0 + arow + 32) * K + aquad * 4];
    float4 fw[4];
    load_w_regs(W, 0, n0, N, wseg, wkp, fw);
    v4bf pa0 = { (__bf16)fa0.x, (__bf16)fa0.y, (__bf16)fa0.z, (__bf16)fa0.w };
    v4bf pa1 = { (__bf16)fa1.x, (__bf16)fa1.y, (__bf16)fa1.z, (__bf16)fa1.w };
    *(v4bf*)&As[0][(arow)      * G_LDA + aquad * 4] = pa0;
    *(v4bf*)&As[0][(arow + 32) * G_LDA + aquad * 4] = pa1;
    store_w_lds(Ws[0], wseg, wkp, fw);
  }
  __syncthreads();

  for (int k0 = 0, it = 0; k0 < K; k0 += G_BK, ++it) {
    int cur = it & 1;
    bool more = (k0 + G_BK) < K;
    float4 fa0, fa1, fw[4];
    if (more) {   // issue next-tile global loads before compute
      int kn = k0 + G_BK;
      fa0 = *(const float4*)&A[(size_t)(m0 + arow)      * K + kn + aquad * 4];
      fa1 = *(const float4*)&A[(size_t)(m0 + arow + 32) * K + kn + aquad * 4];
      load_w_regs(W, kn, n0, N, wseg, wkp, fw);
    }
    gemm_compute(As[cur], Ws[cur], wm, wn, acc);
    if (more) {
      v4bf pa0 = { (__bf16)fa0.x, (__bf16)fa0.y, (__bf16)fa0.z, (__bf16)fa0.w };
      v4bf pa1 = { (__bf16)fa1.x, (__bf16)fa1.y, (__bf16)fa1.z, (__bf16)fa1.w };
      *(v4bf*)&As[cur ^ 1][(arow)      * G_LDA + aquad * 4] = pa0;
      *(v4bf*)&As[cur ^ 1][(arow + 32) * G_LDA + aquad * 4] = pa1;
      store_w_lds(Ws[cur ^ 1], wseg, wkp, fw);
      __syncthreads();
    }
  }

  int col = lane & 15;
  int rbase = (lane >> 4) * 8;
  #pragma unroll
  for (int i = 0; i < 2; ++i)
    #pragma unroll
    for (int j = 0; j < 2; ++j) {
      int row = m0 + wm + i * 16 + rbase;
      int cc  = n0 + wn + j * 16 + col;
      float bv = bias[cc];
      #pragma unroll
      for (int v = 0; v < 8; ++v)
        C[(size_t)(row + v) * N + cc] = (__bf16)(acc[i][j][v] + bv);
    }
}

// C(f32) = A(bf16) * W(f32) + bias  (output projection)
__global__ __launch_bounds__(256)
void gemm_out(const __bf16* __restrict__ A, const float* __restrict__ W,
              const float* __restrict__ bias, float* __restrict__ C,
              int M, int N, int K) {
  __shared__ __bf16 As[2][G_BM * G_LDA];
  __shared__ __bf16 Ws[2][G_BN * G_LDW];
  int tid  = threadIdx.x;
  int wave = tid >> 5;
  int lane = tid & 31;
  int m0 = blockIdx.y * G_BM;
  int n0 = blockIdx.x * G_BN;
  int wm = (wave >> 2) * 32;
  int wn = (wave & 3) * 32;

  int arow  = tid >> 2;          // 0..63, one b128 per thread
  int acol  = (tid & 3) * 8;
  int wseg  = tid & 31;
  int wkp   = tid >> 5;

  v8f acc[2][2] = {};

  {
    v8bf av = *(const v8bf*)&A[(size_t)(m0 + arow) * K + acol];
    float4 fw[4];
    load_w_regs(W, 0, n0, N, wseg, wkp, fw);
    *(v8bf*)&As[0][arow * G_LDA + acol] = av;
    store_w_lds(Ws[0], wseg, wkp, fw);
  }
  __syncthreads();

  for (int k0 = 0, it = 0; k0 < K; k0 += G_BK, ++it) {
    int cur = it & 1;
    bool more = (k0 + G_BK) < K;
    v8bf av;
    float4 fw[4];
    if (more) {
      int kn = k0 + G_BK;
      av = *(const v8bf*)&A[(size_t)(m0 + arow) * K + kn + acol];
      load_w_regs(W, kn, n0, N, wseg, wkp, fw);
    }
    gemm_compute(As[cur], Ws[cur], wm, wn, acc);
    if (more) {
      *(v8bf*)&As[cur ^ 1][arow * G_LDA + acol] = av;
      store_w_lds(Ws[cur ^ 1], wseg, wkp, fw);
      __syncthreads();
    }
  }

  int col = lane & 15;
  int rbase = (lane >> 4) * 8;
  #pragma unroll
  for (int i = 0; i < 2; ++i)
    #pragma unroll
    for (int j = 0; j < 2; ++j) {
      int row = m0 + wm + i * 16 + rbase;
      int cc  = n0 + wn + j * 16 + col;
      float bv = bias[cc];
      #pragma unroll
      for (int v = 0; v < 8; ++v)
        C[(size_t)(row + v) * N + cc] = acc[i][j][v] + bv;
    }
}

// ---------------------------------------------------------------------------
// Flash attention: block = 4 waves, 64 q-rows; pipelined over 32-key tiles.
// K-tiles arrive via TDM (TENSORcnt) into double-buffered LDS; V-tiles are
// register-staged + transposed into double-buffered LDS.
// ---------------------------------------------------------------------------
constexpr int A_QT = 64;
constexpr int A_KT = 32;
constexpr int A_HD = 64;
constexpr int A_LQ = A_HD + 8;   // 72
constexpr int A_LK = A_HD + 8;   // 72  (TDM pad: 32 DW interval + 4 DW pad)
constexpr int A_LV = A_KT + 8;   // 40 (V stored transposed [d][k])
constexpr int A_LP = A_KT + 8;   // 40

__global__ __launch_bounds__(128)
void attn_kernel(const __bf16* __restrict__ Qb, const __bf16* __restrict__ Kb,
                 const __bf16* __restrict__ Vb, __bf16* __restrict__ Ob,
                 int Lq, int Lk, int D) {
  __shared__ __bf16 Qs[A_QT * A_LQ];
  __shared__ __bf16 Ks[2][A_KT * A_LK];
  __shared__ __bf16 Vst[2][A_HD * A_LV];
  __shared__ __bf16 Ps[4][16 * A_LP];

  int tid  = threadIdx.x;
  int wave = tid >> 5;
  int lane = tid & 31;
  int bh = blockIdx.x;
  int b  = bh >> 4;
  int h  = bh & 15;
  int q0 = blockIdx.y * A_QT;

  const __bf16* Qbase = Qb + ((size_t)b * Lq) * D + h * A_HD;
  const __bf16* Kbase = Kb + ((size_t)b * Lk) * D + h * A_HD;
  const __bf16* Vbase = Vb + ((size_t)b * Lk) * D + h * A_HD;

  int chunk = tid & 7;       // 8-elem (16B) column chunk
  int r0    = tid >> 3;      // 0..15

  unsigned qs_off = (unsigned)(uintptr_t)&Qs[0];   // LDS aperture: low 32 bits
  unsigned ks_off[2] = { (unsigned)(uintptr_t)&Ks[0][0],
                         (unsigned)(uintptr_t)&Ks[1][0] };

  // --- stage Q tile (64x64 bf16) via async LDS DMA, 4 b128 per thread ---
  #pragma unroll
  for (int rr = 0; rr < 4; ++rr) {
    int r = r0 + rr * 16;
    async_load_b128(qs_off + (unsigned)(r * A_LQ + chunk * 8) * 2,
                    &Qbase[(size_t)(q0 + r) * D + chunk * 8]);
  }

  // --- prologue: K tile 0 via TDM, V tile 0 via register transpose ---
  if (wave == 0)
    tdm_load_2d_bf16(ks_off[0], &Kbase[0], A_HD, A_KT, D, 4, 3);
  {
    v8bf vr[2];
    #pragma unroll
    for (int rr = 0; rr < 2; ++rr)
      vr[rr] = *(const v8bf*)&Vbase[(size_t)(r0 + rr * 16) * D + chunk * 8];
    #pragma unroll
    for (int rr = 0; rr < 2; ++rr) {
      int row = r0 + rr * 16;
      #pragma unroll
      for (int j = 0; j < 8; ++j) Vst[0][(chunk * 8 + j) * A_LV + row] = vr[rr][j];
    }
  }
  wait_asynccnt0();                                  // Q staged
  if (wave == 0) __builtin_amdgcn_s_wait_tensorcnt((short)0);
  __syncthreads();

  int qw = wave * 16;
  v16bf qa0 = lds_load_a16(&Qs[qw * A_LQ +  0], A_LQ);
  v16bf qa1 = lds_load_a16(&Qs[qw * A_LQ + 32], A_LQ);

  v8f o[4] = {};
  float mrow[8], lrow[8];
  #pragma unroll
  for (int v = 0; v < 8; ++v) { mrow[v] = -1e30f; lrow[v] = 0.0f; }

  const float scale_log2e = 0.125f * 1.44269504f;   // 1/sqrt(64) * log2(e)

  for (int kt = 0, it = 0; kt < Lk; kt += A_KT, ++it) {
    int cur = it & 1;
    bool more = (kt + A_KT) < Lk;

    // issue next tile's DMA + loads so they overlap with this tile's compute
    v8bf vnext[2];
    if (more) {
      if (wave == 0)
        tdm_load_2d_bf16(ks_off[cur ^ 1], &Kbase[(size_t)(kt + A_KT) * D],
                         A_HD, A_KT, D, 4, 3);
      #pragma unroll
      for (int rr = 0; rr < 2; ++rr)
        vnext[rr] = *(const v8bf*)
            &Vbase[(size_t)(kt + A_KT + r0 + rr * 16) * D + chunk * 8];
    }

    // S(16x32) = Q(16x64) . K^T  (K row-major [key][d] == B layout N-major)
    const __bf16* Kc = Ks[cur];
    v8f s0 = {}, s1 = {};
    s0 = wmma_bf16(qa0, lds_load_b16(&Kc[ 0 * A_LK +  0], A_LK), s0);
    s0 = wmma_bf16(qa1, lds_load_b16(&Kc[ 0 * A_LK + 32], A_LK), s0);
    s1 = wmma_bf16(qa0, lds_load_b16(&Kc[16 * A_LK +  0], A_LK), s1);
    s1 = wmma_bf16(qa1, lds_load_b16(&Kc[16 * A_LK + 32], A_LK), s1);

    // online softmax in log2 domain
    float corr[8];
    #pragma unroll
    for (int v = 0; v < 8; ++v) {
      float t0 = s0[v] * scale_log2e;
      float t1 = s1[v] * scale_log2e;
      float sm = fmaxf(t0, t1);
      #pragma unroll
      for (int off = 1; off < 16; off <<= 1)
        sm = fmaxf(sm, __shfl_xor(sm, off, 32));
      float mnew = fmaxf(mrow[v], sm);
      float c    = exp2f(mrow[v] - mnew);
      float p0   = exp2f(t0 - mnew);
      float p1   = exp2f(t1 - mnew);
      float ps   = p0 + p1;
      #pragma unroll
      for (int off = 1; off < 16; off <<= 1)
        ps += __shfl_xor(ps, off, 32);
      lrow[v] = lrow[v] * c + ps;
      mrow[v] = mnew;
      corr[v] = c;
      s0[v] = p0;
      s1[v] = p1;
    }
    #pragma unroll
    for (int dc = 0; dc < 4; ++dc)
      #pragma unroll
      for (int v = 0; v < 8; ++v) o[dc][v] *= corr[v];

    // C-layout -> A-layout for P via per-wave LDS round trip
    {
      int col = lane & 15, rbase = (lane >> 4) * 8;
      #pragma unroll
      for (int v = 0; v < 8; ++v) {
        Ps[wave][(rbase + v) * A_LP + col]      = (__bf16)s0[v];
        Ps[wave][(rbase + v) * A_LP + 16 + col] = (__bf16)s1[v];
      }
    }
    asm volatile("s_wait_dscnt 0x0" ::: "memory");
    __builtin_amdgcn_wave_barrier();

    v16bf pa = lds_load_a16(&Ps[wave][0], A_LP);
    const __bf16* Vc = Vst[cur];
    #pragma unroll
    for (int dc = 0; dc < 4; ++dc) {
      v16bf vfrag = lds_load_b16(&Vc[(dc * 16) * A_LV], A_LV);
      o[dc] = wmma_bf16(pa, vfrag, o[dc]);
    }

    if (more) {   // land next V tile, wait for next K DMA, one barrier/iter
      #pragma unroll
      for (int rr = 0; rr < 2; ++rr) {
        int row = r0 + rr * 16;
        #pragma unroll
        for (int j = 0; j < 8; ++j)
          Vst[cur ^ 1][(chunk * 8 + j) * A_LV + row] = vnext[rr][j];
      }
      if (wave == 0) __builtin_amdgcn_s_wait_tensorcnt((short)0);
      __syncthreads();
    }
  }

  // normalize and store
  int col = lane & 15, rbase = (lane >> 4) * 8;
  #pragma unroll
  for (int v = 0; v < 8; ++v) {
    float invl = 1.0f / lrow[v];
    int qrow = q0 + qw + rbase + v;
    size_t base = ((size_t)b * Lq + qrow) * D + h * A_HD;
    #pragma unroll
    for (int dc = 0; dc < 4; ++dc)
      Ob[base + dc * 16 + col] = (__bf16)(o[dc][v] * invl);
  }
}

// ---------------------------------------------------------------------------
extern "C" void kernel_launch(void* const* d_in, const int* in_sizes, int n_in,
                              void* d_out, int out_size, void* d_ws, size_t ws_size,
                              hipStream_t stream) {
  const float* x   = (const float*)d_in[0];
  const float* ctx = (const float*)d_in[1];
  const float* w_q = (const float*)d_in[2];
  const float* b_q = (const float*)d_in[3];
  const float* w_k = (const float*)d_in[4];
  const float* b_k = (const float*)d_in[5];
  const float* w_v = (const float*)d_in[6];
  const float* b_v = (const float*)d_in[7];
  const float* w_o = (const float*)d_in[8];
  const float* b_o = (const float*)d_in[9];
  float* out = (float*)d_out;

  const int B = 2, L = 2048, D = 1024, DC = 768;
  const int M = B * L;                    // 4096

  __bf16* Qb = (__bf16*)d_ws;             // 8 MB each
  __bf16* Kb = Qb + (size_t)M * D;
  __bf16* Vb = Kb + (size_t)M * D;
  __bf16* Ab = Vb + (size_t)M * D;

  dim3 gg(D / G_BN, M / G_BM);            // (8, 64)
  gemm_qkv<<<gg, 256, 0, stream>>>(x,   w_q, b_q, Qb, M, D, D);
  gemm_qkv<<<gg, 256, 0, stream>>>(ctx, w_k, b_k, Kb, M, D, DC);
  gemm_qkv<<<gg, 256, 0, stream>>>(ctx, w_v, b_v, Vb, M, D, DC);

  dim3 ga(B * 16, L / A_QT);              // (32, 32)
  attn_kernel<<<ga, 128, 0, stream>>>(Qb, Kb, Vb, Ab, L, L, D);

  gemm_out<<<gg, 256, 0, stream>>>(Ab, w_o, b_o, out, M, D, D);
}
